// SelfAttention_16475494547604
// MI455X (gfx1250) — matrix operand order repaired and checked by reading, not compiled, not run on hardware
//
#include <hip/hip_runtime.h>
#include <hip/hip_bf16.h>
#include <math.h>

// Problem constants (from reference): B=64, T=320, H=256, D=2H=512
#define B_   64
#define T_   320
#define H_   256
#define D_   512
#define M_   (B_*T_)    // 20480 flattened (b,t) rows
#define G3H  (3*H_)     // 768
#define D2   (2*D_)     // 1024

typedef __bf16 bf16;
typedef __attribute__((ext_vector_type(16))) __bf16 v16bf;
typedef __attribute__((ext_vector_type(8)))  __bf16 v8bf;
typedef __attribute__((ext_vector_type(8)))  float  v8f;

__device__ __forceinline__ float sigmoidf_(float x) { return 1.0f / (1.0f + __expf(-x)); }

// CDNA5 async global->LDS copy (16B per lane), tracked by ASYNCcnt.
__device__ __forceinline__ void async_cp16(unsigned lds_byte_off, const void* g) {
  asm volatile("global_load_async_to_lds_b128 %0, %1, off"
               :: "v"(lds_byte_off), "v"(g) : "memory");
}

// ---------------------------------------------------------------------------
// f32 -> bf16 conversion
// ---------------------------------------------------------------------------
__global__ void f32_to_bf16_kernel(const float* __restrict__ in, bf16* __restrict__ out, int n) {
  int i = blockIdx.x * blockDim.x + threadIdx.x;
  if (i < n) out[i] = (bf16)in[i];
}

__global__ void zero_kernel(float* __restrict__ p, int n) {
  int i = blockIdx.x * blockDim.x + threadIdx.x;
  if (i < n) p[i] = 0.0f;
}

// ---------------------------------------------------------------------------
// WMMA bf16 GEMM:  out[m,n] = sum_k A[m,k] * W[n,k]  (+ bias[n])
// mode 0: outF[m*N+n] = acc + bias[n]                        (f32 store)
// mode 1: g = sigmoid(acc + bias[n]); pg = [v | cell];
//         outB[m*N+n] = bf16(pg[m,n] * g)                    (gate epilogue)
// Macro tile 64(M) x 128(N) x 32(K); 8 waves; each wave does 2x2 16x16 tiles.
// Global->LDS via async-to-LDS with ping-pong double buffering (ASYNCcnt).
// ---------------------------------------------------------------------------
__launch_bounds__(256)
__global__ void gemm_bf16_kernel(const bf16* __restrict__ A, const bf16* __restrict__ W,
                                 const float* __restrict__ bias,
                                 float* __restrict__ outF, bf16* __restrict__ outB,
                                 const float* __restrict__ epiV, const float* __restrict__ epiCell,
                                 int M, int N, int K, int mode)
{
  __shared__ bf16 sA[2][64 * 40];    // 64 rows x 32 K, stride 40 (pad vs banks)
  __shared__ bf16 sB[2][128 * 40];   // 128 N-rows x 32 K

  const int tid   = threadIdx.x;
  const int wave  = tid >> 5;
  const int lane  = tid & 31;
  const int half  = (lane >> 4) & 1;   // wave32 half
  const int l16   = lane & 15;
  const int waveM = wave & 1;          // 0..1 -> 32 rows each
  const int waveN = wave >> 1;         // 0..3 -> 32 cols each
  const int m0 = blockIdx.y * 64;
  const int n0 = blockIdx.x * 128;

  v8f acc[2][2] = {};

  const int arow = tid >> 2, aseg = tid & 3;       // A: 256 8-elem segments
  const int br0 = tid >> 2,         bg0 = tid & 3; // B: 512 segments, 2/thread
  const int br1 = (tid + 256) >> 2, bg1 = (tid + 256) & 3;

  // Issue one stage (3 async 16B transfers per thread) into buffer `buf`.
  auto stage = [&](int buf, int k0) {
    async_cp16((unsigned)(uintptr_t)&sA[buf][arow * 40 + aseg * 8],
               &A[(size_t)(m0 + arow) * K + k0 + aseg * 8]);
    async_cp16((unsigned)(uintptr_t)&sB[buf][br0 * 40 + bg0 * 8],
               &W[(size_t)(n0 + br0) * K + k0 + bg0 * 8]);
    async_cp16((unsigned)(uintptr_t)&sB[buf][br1 * 40 + bg1 * 8],
               &W[(size_t)(n0 + br1) * K + k0 + bg1 * 8]);
  };

  stage(0, 0);
  int buf = 0;
  for (int k0 = 0; k0 < K; k0 += 32) {
    const bool hasNext = (k0 + 32 < K);
    if (hasNext) stage(buf ^ 1, k0 + 32);   // prefetch next tile into other buffer
    // Async loads complete in order: <=3 outstanding drains the previous stage.
    if (hasNext) asm volatile("s_wait_asynccnt 0x3" ::: "memory");
    else         asm volatile("s_wait_asynccnt 0x0" ::: "memory");
    __syncthreads();

    // A fragment (16x32 bf16): lanes 0-15 row=l16 K={0..7,16..23}; lanes 16-31 K={8..15,24..31}
    v16bf afrag[2], bfrag[2];
#pragma unroll
    for (int mi = 0; mi < 2; ++mi) {
      int row = waveM * 32 + mi * 16 + l16;
      union { v16bf v; v8bf h[2]; } u;
      u.h[0] = *(const v8bf*)&sA[buf][row * 40 + half * 8];
      u.h[1] = *(const v8bf*)&sA[buf][row * 40 + 16 + half * 8];
      afrag[mi] = u.v;
    }
    // B fragment (32x16 bf16): lanes 0-15 K={0..7,16..23}; lanes 16-31 K={8..15,24..31}, N=l16
#pragma unroll
    for (int ni = 0; ni < 2; ++ni) {
      int col = waveN * 32 + ni * 16 + l16;
      union { v16bf v; v8bf h[2]; } u;
      u.h[0] = *(const v8bf*)&sB[buf][col * 40 + half * 8];
      u.h[1] = *(const v8bf*)&sB[buf][col * 40 + 16 + half * 8];
      bfrag[ni] = u.v;
    }
#pragma unroll
    for (int mi = 0; mi < 2; ++mi)
#pragma unroll
      for (int ni = 0; ni < 2; ++ni)
        acc[mi][ni] = __builtin_amdgcn_wmma_f32_16x16x32_bf16(
            false, afrag[mi], false, bfrag[ni], (short)0, acc[mi][ni], false, false);
    __syncthreads();   // all waves done reading `buf` before it is re-staged
    buf ^= 1;
  }

  // Epilogue. C/D layout: lane l, reg r -> M = r + (l>=16?8:0), N = l&15
#pragma unroll
  for (int mi = 0; mi < 2; ++mi) {
#pragma unroll
    for (int ni = 0; ni < 2; ++ni) {
      int nCol = n0 + waveN * 32 + ni * 16 + l16;
      float bv = bias ? bias[nCol] : 0.0f;
#pragma unroll
      for (int r = 0; r < 8; ++r) {
        int m = m0 + waveM * 32 + mi * 16 + half * 8 + r;
        float val = acc[mi][ni][r] + bv;
        if (mode == 0) {
          outF[(size_t)m * N + nCol] = val;
        } else {
          float g   = sigmoidf_(val);
          float pgv = (nCol < D_) ? epiV[(size_t)m * D_ + nCol]
                                  : epiCell[(size_t)m * D_ + (nCol - D_)];
          outB[(size_t)m * (size_t)N + nCol] = (bf16)(pgv * g);
        }
      }
    }
  }
}

// ---------------------------------------------------------------------------
// Attention: per (b,tq) block (320 threads, one tk per thread):
//   scores[tk] = sum_d cw[d] * tanh(cwm[d] + cm[b,tk,d]); softmax; cell = attn @ v
// ---------------------------------------------------------------------------
__launch_bounds__(320)
__global__ void attn_kernel(const float* __restrict__ v, const float* __restrict__ cm,
                            const float* __restrict__ cwm, float* __restrict__ cell)
{
  __shared__ float s_cw[D_];
  __shared__ float s_cwm[D_];
  __shared__ float s_p[T_];
  __shared__ float s_part[10];
  __shared__ float s_bc;

  const int bq  = blockIdx.x;        // b*T + tq
  const int b   = bq / T_;
  const int tid = threadIdx.x;

  for (int i = tid; i < D_; i += 320) {
    s_cw[i]  = v[(size_t)bq * D_ + i];
    s_cwm[i] = cwm[(size_t)bq * D_ + i];
  }
  __syncthreads();

  const int tk = tid;                // 0..319
  const float* cmrow = cm + ((size_t)b * T_ + tk) * D_;
  float sc = 0.0f;
  for (int d = 0; d < D_; ++d)
    sc += s_cw[d] * tanhf(s_cwm[d] + cmrow[d]);

  // block max (wave32 shuffle + 10-wave LDS combine)
  float w = sc;
  for (int o = 16; o > 0; o >>= 1) w = fmaxf(w, __shfl_xor(w, o, 32));
  if ((tid & 31) == 0) s_part[tid >> 5] = w;
  __syncthreads();
  if (tid == 0) {
    float mx = s_part[0];
    for (int i = 1; i < 10; ++i) mx = fmaxf(mx, s_part[i]);
    s_bc = mx;
  }
  __syncthreads();
  float mx = s_bc;
  float p  = __expf(sc - mx);
  s_p[tk] = p;
  __syncthreads();
  // block sum
  float ws = p;
  for (int o = 16; o > 0; o >>= 1) ws += __shfl_xor(ws, o, 32);
  if ((tid & 31) == 0) s_part[tid >> 5] = ws;
  __syncthreads();
  if (tid == 0) {
    float sm = 0.0f;
    for (int i = 0; i < 10; ++i) sm += s_part[i];
    s_bc = 1.0f / sm;
  }
  __syncthreads();
  float inv = s_bc;

  // cell[d] = (attn @ v)[d]  -- coalesced over d
  for (int d = tid; d < D_; d += 320) {
    float a = 0.0f;
    for (int t = 0; t < T_; ++t)
      a += s_p[t] * v[((size_t)b * T_ + t) * D_ + d];
    cell[(size_t)bq * D_ + d] = a * inv;
  }
}

// ---------------------------------------------------------------------------
// pg = concat(cw, cell) -> bf16  (input to the Wg GEMM)
// ---------------------------------------------------------------------------
__global__ void pg_build_kernel(const float* __restrict__ v, const float* __restrict__ cell,
                                bf16* __restrict__ pg)
{
  size_t i = (size_t)blockIdx.x * blockDim.x + threadIdx.x;
  if (i >= (size_t)M_ * D2) return;
  int n    = (int)(i & (D2 - 1));
  size_t m = i >> 10;
  float val = (n < D_) ? v[m * D_ + n] : cell[m * D_ + (n - D_)];
  pg[i] = (bf16)val;
}

// ---------------------------------------------------------------------------
// GRU recurrence (sequential over t). S1: gh = h @ Whh^T + bhh (both dirs)
// ---------------------------------------------------------------------------
__global__ void gru_gh_kernel(const float* __restrict__ hf, const float* __restrict__ hb,
                              const float* __restrict__ Whhf, const float* __restrict__ Whhb,
                              const float* __restrict__ bhhf, const float* __restrict__ bhhb,
                              float* __restrict__ gh)
{
  int idx = blockIdx.x * blockDim.x + threadIdx.x;   // 0 .. 2*B*3H-1
  if (idx >= 2 * B_ * G3H) return;
  int dir = idx / (B_ * G3H);
  int rem = idx - dir * (B_ * G3H);
  int b = rem / G3H, j = rem - b * G3H;
  const float* h   = dir ? hb   : hf;
  const float* Whh = dir ? Whhb : Whhf;
  const float* bhh = dir ? bhhb : bhhf;
  float a = bhh[j];
  const float* hr = h + b * H_;
  const float* wr = Whh + (size_t)j * H_;
  for (int k = 0; k < H_; ++k) a = fmaf(hr[k], wr[k], a);
  gh[idx] = a;
}

// S2: gate combine + hidden update + output write (torch GRUCell gate order r,z,n)
__global__ void gru_update_kernel(const float* __restrict__ gxf, const float* __restrict__ gxb,
                                  const float* __restrict__ gh,
                                  float* __restrict__ hf, float* __restrict__ hb,
                                  float* __restrict__ out, int t)
{
  int idx = blockIdx.x * blockDim.x + threadIdx.x;   // 0 .. 2*B*H-1
  if (idx >= 2 * B_ * H_) return;
  int dir = idx / (B_ * H_);
  int rem = idx - dir * (B_ * H_);
  int b = rem / H_, j = rem - b * H_;
  const float* gx = dir ? gxb : gxf;
  float* h = dir ? hb : hf;
  size_t m = (size_t)b * T_ + t;
  float i_r = gx[m * G3H + j];
  float i_z = gx[m * G3H + H_ + j];
  float i_n = gx[m * G3H + 2 * H_ + j];
  const float* ghd = gh + dir * (B_ * G3H) + b * G3H;
  float h_r = ghd[j], h_z = ghd[H_ + j], h_n = ghd[2 * H_ + j];
  float r = sigmoidf_(i_r + h_r);
  float z = sigmoidf_(i_z + h_z);
  float n = tanhf(i_n + r * h_n);
  float hprev = h[b * H_ + j];
  float hnew = (1.0f - z) * n + z * hprev;
  h[b * H_ + j] = hnew;
  out[((size_t)t * B_ + b) * (2 * H_) + dir * H_ + j] = hnew;
}

// ---------------------------------------------------------------------------
extern "C" void kernel_launch(void* const* d_in, const int* in_sizes, int n_in,
                              void* d_out, int out_size, void* d_ws, size_t ws_size,
                              hipStream_t stream)
{
  const float* v    = (const float*)d_in[0];
  const float* W1   = (const float*)d_in[1];
  const float* b1   = (const float*)d_in[2];
  const float* W2   = (const float*)d_in[3];
  const float* b2   = (const float*)d_in[4];
  const float* Wg   = (const float*)d_in[5];
  const float* bg   = (const float*)d_in[6];
  const float* Wihf = (const float*)d_in[7];
  const float* Whhf = (const float*)d_in[8];
  const float* bihf = (const float*)d_in[9];
  const float* bhhf = (const float*)d_in[10];
  const float* Wihb = (const float*)d_in[11];
  const float* Whhb = (const float*)d_in[12];
  const float* bihb = (const float*)d_in[13];
  const float* bhhb = (const float*)d_in[14];
  float* out = (float*)d_out;

  char* base = (char*)d_ws;
  size_t off = 0;
  auto alloc = [&](size_t bytes) -> char* {
    char* p = base + off;
    off = (off + bytes + 255) & ~(size_t)255;
    return p;
  };

  bf16* v_bf    = (bf16*)alloc((size_t)M_ * D_ * 2);
  bf16* W1_bf   = (bf16*)alloc((size_t)D_ * D_ * 2);
  bf16* W2_bf   = (bf16*)alloc((size_t)D_ * D_ * 2);
  bf16* Wg_bf   = (bf16*)alloc((size_t)D2 * D2 * 2);
  bf16* Wihf_bf = (bf16*)alloc((size_t)G3H * D2 * 2);
  bf16* Wihb_bf = (bf16*)alloc((size_t)G3H * D2 * 2);
  float* cm     = (float*)alloc((size_t)M_ * D_ * 4);
  float* cwm    = (float*)alloc((size_t)M_ * D_ * 4);
  float* cell   = (float*)alloc((size_t)M_ * D_ * 4);
  bf16*  pg_bf  = (bf16*)alloc((size_t)M_ * D2 * 2);
  bf16*  gi_bf  = (bf16*)alloc((size_t)M_ * D2 * 2);
  float* gx_f   = (float*)alloc((size_t)M_ * G3H * 4);
  float* gx_b   = (float*)alloc((size_t)M_ * G3H * 4);
  float* gh     = (float*)alloc((size_t)2 * B_ * G3H * 4);
  float* h_f    = (float*)alloc((size_t)B_ * H_ * 4);
  float* h_b    = (float*)alloc((size_t)B_ * H_ * 4);
  (void)ws_size; (void)in_sizes; (void)n_in; (void)out_size;

  auto cvt = [&](const float* src, bf16* dst, int n) {
    f32_to_bf16_kernel<<<(n + 255) / 256, 256, 0, stream>>>(src, dst, n);
  };
  // 1) bf16 copies of activations + weights for the WMMA GEMMs
  cvt(v,    v_bf,    M_ * D_);
  cvt(W1,   W1_bf,   D_ * D_);
  cvt(W2,   W2_bf,   D_ * D_);
  cvt(Wg,   Wg_bf,   D2 * D2);
  cvt(Wihf, Wihf_bf, G3H * D2);
  cvt(Wihb, Wihb_bf, G3H * D2);

  // 2) cm = v @ W2^T + b2   (hoisted; identical every step)
  gemm_bf16_kernel<<<dim3(D_ / 128, M_ / 64), 256, 0, stream>>>(
      v_bf, W2_bf, b2, cm, nullptr, nullptr, nullptr, M_, D_, D_, 0);
  // 3) cwm = v @ W1^T + b1  (all timesteps at once)
  gemm_bf16_kernel<<<dim3(D_ / 128, M_ / 64), 256, 0, stream>>>(
      v_bf, W1_bf, b1, cwm, nullptr, nullptr, nullptr, M_, D_, D_, 0);

  // 4) attention (tanh-score + softmax + context) for all (b,t)
  attn_kernel<<<M_, 320, 0, stream>>>(v, cm, cwm, cell);

  // 5) pg = [cw | cell] in bf16
  pg_build_kernel<<<(M_ * D2) / 256, 256, 0, stream>>>(v, cell, pg_bf);

  // 6) gate GEMM + fused sigmoid*pg epilogue -> gi (bf16)
  gemm_bf16_kernel<<<dim3(D2 / 128, M_ / 64), 256, 0, stream>>>(
      pg_bf, Wg_bf, bg, nullptr, gi_bf, v, cell, M_, D2, D2, 1);

  // 7) gx = gi @ Wih^T + bih  for both directions (hoisted out of recurrence)
  gemm_bf16_kernel<<<dim3(G3H / 128, M_ / 64), 256, 0, stream>>>(
      gi_bf, Wihf_bf, bihf, gx_f, nullptr, nullptr, nullptr, M_, G3H, D2, 0);
  gemm_bf16_kernel<<<dim3(G3H / 128, M_ / 64), 256, 0, stream>>>(
      gi_bf, Wihb_bf, bihb, gx_b, nullptr, nullptr, nullptr, M_, G3H, D2, 0);

  // 8) h0 = 0
  zero_kernel<<<(B_ * H_ + 255) / 256, 256, 0, stream>>>(h_f, B_ * H_);
  zero_kernel<<<(B_ * H_ + 255) / 256, 256, 0, stream>>>(h_b, B_ * H_);

  // 9) sequential GRU recurrence: only h @ Whh^T + gate math per step
  for (int t = 0; t < T_; ++t) {
    gru_gh_kernel<<<(2 * B_ * G3H + 255) / 256, 256, 0, stream>>>(
        h_f, h_b, Whhf, Whhb, bhhf, bhhb, gh);
    gru_update_kernel<<<(2 * B_ * H_ + 255) / 256, 256, 0, stream>>>(
        gx_f, gx_b, gh, h_f, h_b, out, t);
  }
}
